// CNNQNetwork_24833500905805
// MI455X (gfx1250) — compile-verified, matrix operand order checked by reading, not run on hardware
//
#include <hip/hip_runtime.h>
#include <hip/hip_bf16.h>

// CDNA5 / gfx1250: wave32, WMMA 16x16x32 f16 -> f32
typedef __attribute__((ext_vector_type(16))) _Float16 v16h;
typedef __attribute__((ext_vector_type(8)))  float    v8f;

union V16 { v16h v; _Float16 e[16]; };
union V8  { v8f  v; float    e[8];  };

__device__ __forceinline__ v8f wmma16x16x32(v16h a, v16h b, v8f c) {
  // D = A(16x32,f16) * B(32x16,f16) + C(16x16,f32)
  return __builtin_amdgcn_wmma_f32_16x16x32_f16(false, a, false, b, (short)0, c, false, false);
}

#define GN_EPS 1e-5f

// ---------------------------------------------------------------------------
// Weight prep -> f16 WMMA B-operand tiles.
// B-operand layout per (ktile, ntile): lane = (n&15) + 16*khalf, elem i -> K.
// Conv GEMM K-ORDER IS kk-MAJOR:  k = kk*Cin + cin   (kk = kernel tap 0/1).
// This makes the 16 A elements of any (lane, ktile) contiguous in cin, so the
// A operand is a single 32B LDS vector load. A and B use the same K order, so
// the contraction is unchanged.
// ---------------------------------------------------------------------------

// FC weights: w is (N=256, K=7424) row-major; K order = natural feature order.
__global__ void prep_fc(const float* __restrict__ w, _Float16* __restrict__ dst,
                        int ntOff, int N, int K) {
  int e = blockIdx.x * blockDim.x + threadIdx.x;
  if (e >= N * K) return;
  int n = e / K, k = e - n * K;
  int kt = k >> 5, kh = (k >> 4) & 1, i = k & 15;
  int nt = ntOff + (n >> 4);
  int lane = (n & 15) + (kh << 4);
  dst[(((unsigned)(kt * 32 + nt) * 32u + (unsigned)lane) << 4) + (unsigned)i] = (_Float16)w[e];
}

// Conv weights: w is (Co=128, Ci, kA, kB), kA*kB==2, flat e = (co*Ci+cin)*2+kk.
// K order: k = kk*Ci + cin.  dst has NT=8 n-tiles.
__global__ void prep_conv(const float* __restrict__ w, _Float16* __restrict__ dst, int Ci) {
  int e = blockIdx.x * blockDim.x + threadIdx.x;
  if (e >= 128 * Ci * 2) return;
  int co  = e / (Ci * 2);
  int rem = e - co * (Ci * 2);
  int cin = rem >> 1, kk = rem & 1;
  int k = kk * Ci + cin;
  int kt = k >> 5, kh = (k >> 4) & 1, i = k & 15;
  int nt = co >> 4;
  int lane = (co & 15) + (kh << 4);
  dst[(((unsigned)(kt * 8 + nt) * 32u + (unsigned)lane) << 4) + (unsigned)i] = (_Float16)w[e];
}

// ---------------------------------------------------------------------------
// Conv + GroupNorm(1) + ReLU. One workgroup = 16 samples, 8 waves = 8 n-tiles.
// LDS activation layout: [s][h][w][c]  (channel innermost -> contiguous A).
// A-operand: lane m = lane&15 (sample), khalf = lane>>4; elem i -> K.
// D layout: VGPR r -> sample = r + 8*(lane>>4), channel col = lane&15.
// ---------------------------------------------------------------------------

// Layer 1: Cin=16, K=32 (single K-tile). khalf == kk directly.
__device__ __forceinline__ void conv1_block(
    const _Float16* __restrict__ xs /*[s][4][4][16]*/, _Float16* __restrict__ dst /*[s][P][128]*/,
    const _Float16* __restrict__ Wp, const float* __restrict__ bias,
    int vertical, int H, int W,
    float* __restrict__ psum_s, float* __restrict__ psq_s, int wid, int lane) {
  const int m = lane & 15, kh2 = lane >> 4;   // kh2 == kk for layer 1
  const int co = wid * 16 + (lane & 15);
  const float bv = bias[co];
  float psum[8], psq[8];
#pragma unroll
  for (int r = 0; r < 8; r++) { psum[r] = 0.f; psq[r] = 0.f; }

  v16h b = *(const v16h*)(Wp + (((unsigned)(wid * 32 + lane)) << 4));
  const int P = H * W;
  for (int p = 0; p < P; p++) {
    int oh = p / W, ow = p - oh * W;
    int hh = vertical ? (oh + kh2) : oh;
    int ww = vertical ? ow : (ow + kh2);
    v16h a = *(const v16h*)(xs + (((m * 4 + hh) * 4 + ww) << 4));  // 16 contiguous cin
    v8f cz = {};
    V8 cu; cu.v = wmma16x16x32(a, b, cz);
#pragma unroll
    for (int r = 0; r < 8; r++) {
      float y = cu.e[r] + bv;
      psum[r] += y; psq[r] += y * y;
      int s = r + (kh2 << 3);
      dst[(unsigned)(s * P + p) * 128u + (unsigned)co] = (_Float16)y;
    }
  }
#pragma unroll
  for (int r = 0; r < 8; r++) {
    for (int msk = 1; msk < 16; msk <<= 1) {
      psum[r] += __shfl_xor(psum[r], msk, 16);
      psq[r]  += __shfl_xor(psq[r],  msk, 16);
    }
  }
  int g = lane & 15;
  if (g < 8) {                                   // deterministic per-wave partials
    psum_s[wid * 16 + g + (kh2 << 3)] = psum[g];
    psq_s [wid * 16 + g + (kh2 << 3)] = psq[g];
  }
}

// Layer 2: Cin=128, K=256 = 8 K-tiles. kt -> kk = kt>>2, cinBase = (kt&3)*32 + khalf*16.
__device__ __forceinline__ void conv2_block(
    const _Float16* __restrict__ src /*[s][srcH][srcW][128]*/, _Float16* __restrict__ ybuf /*[s][P][128]*/,
    const _Float16* __restrict__ Wp, const float* __restrict__ bias,
    int vertical, int srcW, int oH, int oW,
    float* __restrict__ psum_s, float* __restrict__ psq_s, int wid, int lane) {
  const int m = lane & 15, kh2 = lane >> 4;
  const int co = wid * 16 + (lane & 15);
  const float bv = bias[co];
  float psum[8], psq[8];
#pragma unroll
  for (int r = 0; r < 8; r++) { psum[r] = 0.f; psq[r] = 0.f; }

  v16h bt[8];
#pragma unroll
  for (int kt = 0; kt < 8; kt++)
    bt[kt] = *(const v16h*)(Wp + (((unsigned)((kt * 8 + wid) * 32 + lane)) << 4));

  const int P = oH * oW;
  for (int p = 0; p < P; p++) {
    int oh = p / oW, ow = p - oh * oW;
    V8 cu; { v8f z = {}; cu.v = z; }
#pragma unroll
    for (int kt = 0; kt < 8; kt++) {
      int kk = kt >> 2;
      int cinBase = ((kt & 3) << 5) + (kh2 << 4);
      int hh = vertical ? (oh + kk) : oh;
      int ww = vertical ? ow : (ow + kk);
      v16h a = *(const v16h*)(src + (unsigned)(m * 1536) +
                              (unsigned)((hh * srcW + ww) << 7) + (unsigned)cinBase);
      cu.v = wmma16x16x32(a, bt[kt], cu.v);
    }
#pragma unroll
    for (int r = 0; r < 8; r++) {
      float y = cu.e[r] + bv;
      psum[r] += y; psq[r] += y * y;
      int s = r + (kh2 << 3);
      ybuf[(unsigned)(s * P + p) * 128u + (unsigned)co] = (_Float16)y;
    }
  }
#pragma unroll
  for (int r = 0; r < 8; r++) {
    for (int msk = 1; msk < 16; msk <<= 1) {
      psum[r] += __shfl_xor(psum[r], msk, 16);
      psq[r]  += __shfl_xor(psq[r],  msk, 16);
    }
  }
  int g = lane & 15;
  if (g < 8) {
    psum_s[wid * 16 + g + (kh2 << 3)] = psum[g];
    psq_s [wid * 16 + g + (kh2 << 3)] = psq[g];
  }
}

// GroupNorm(1)+ReLU in place on [s][P][128] LDS buffer; scatter features to
// global in WMMA A-operand layout. Iterates feature-major so adjacent lanes
// write adjacent feat halves (coalesced global stores).
__device__ __forceinline__ void gn_relu_feat(
    _Float16* __restrict__ ybuf, int C /*128*/, int P,
    const float* __restrict__ g, const float* __restrict__ gb,
    const float* __restrict__ psum_s, const float* __restrict__ psq_s,
    float* __restrict__ smean, float* __restrict__ sinv,
    _Float16* __restrict__ feat, unsigned tile, int featOff, int tid) {
  __syncthreads();                       // conv partials + y stores visible
  if (tid < 16) {
    float sm = 0.f, sq = 0.f;
#pragma unroll
    for (int w = 0; w < 8; w++) { sm += psum_s[w * 16 + tid]; sq += psq_s[w * 16 + tid]; }
    float inv_n = 1.0f / (float)(C * P);
    float mu = sm * inv_n;
    float var = sq * inv_n - mu * mu;
    smean[tid] = mu;
    sinv[tid] = rsqrtf(var + GN_EPS);
  }
  __syncthreads();
  const int CP = C * P;
  const int total = 16 * CP;
  for (int idx = tid; idx < total; idx += 256) {
    int s = idx / CP;
    int f = idx - s * CP;                // feature index within block = c*P + p
    int c = f / P;
    int p = f - c * P;
    unsigned li = (unsigned)(s * P + p) * 128u + (unsigned)c;
    float y = (float)ybuf[li];
    float yn = (y - smean[s]) * sinv[s] * g[c] + gb[c];
    yn = yn > 0.f ? yn : 0.f;
    ybuf[li] = (_Float16)yn;
    int fg = featOff + f;                // block offsets are multiples of 32
    unsigned addr = ((tile * 232u + (unsigned)(fg >> 5)) * 32u +
                     (unsigned)(s + (((fg >> 4) & 1) << 4))) * 16u + (unsigned)(fg & 15);
    feat[addr] = (_Float16)yn;
  }
  __syncthreads();
}

__global__ __launch_bounds__(256) void conv_gn_kernel(
    const float* __restrict__ x,
    const float* b_h1, const float* b_v1, const float* b_hh,
    const float* b_hv, const float* b_vh, const float* b_vv,
    const float* gw_h1, const float* gb_h1, const float* gw_v1, const float* gb_v1,
    const float* gw_hh, const float* gb_hh, const float* gw_hv, const float* gb_hv,
    const float* gw_vh, const float* gb_vh, const float* gw_vv, const float* gb_vv,
    const _Float16* __restrict__ wh1, const _Float16* __restrict__ wv1,
    const _Float16* __restrict__ whh, const _Float16* __restrict__ whv,
    const _Float16* __restrict__ wvh, const _Float16* __restrict__ wvv,
    _Float16* __restrict__ feat) {
  extern __shared__ char smem[];
  _Float16* xs   = (_Float16*)smem;        // [16][4][4][16]   4096 halfs
  _Float16* h1s  = xs  + 4096;             // [16][4][3][128] 24576
  _Float16* v1s  = h1s + 24576;            // [16][3][4][128] 24576
  _Float16* y2s  = v1s + 24576;            // [16][P<=9][128] 18432
  float* psum_s  = (float*)(y2s + 18432);  // [8][16]
  float* psq_s   = psum_s + 128;           // [8][16]
  float* smean   = psq_s + 128;            // [16]
  float* sinv    = smean + 16;             // [16]

  const int tid = threadIdx.x;
  const int lane = tid & 31, wid = tid >> 5;
  const unsigned tile = blockIdx.x;

  // load x: global [s][cin][h][w] -> LDS [s][h][w][cin]
  for (int idx = tid; idx < 4096; idx += 256) {
    int s = idx >> 8, r = idx & 255;
    int cin = r >> 4, hw = r & 15;          // hw = h*4+w
    xs[(unsigned)((s << 4) + hw) * 16u + (unsigned)cin] =
        (_Float16)x[(size_t)tile * 4096u + (unsigned)idx];
  }
  __syncthreads();

  // layer 1 (feat offsets: h1=0, v1=1536)
  conv1_block(xs, h1s, wh1, b_h1, 0, 4, 3, psum_s, psq_s, wid, lane);
  gn_relu_feat(h1s, 128, 12, gw_h1, gb_h1, psum_s, psq_s, smean, sinv, feat, tile, 0, tid);
  conv1_block(xs, v1s, wv1, b_v1, 1, 3, 4, psum_s, psq_s, wid, lane);
  gn_relu_feat(v1s, 128, 12, gw_v1, gb_v1, psum_s, psq_s, smean, sinv, feat, tile, 1536, tid);

  // layer 2 (feat offsets: hh=3072, hv=4096, vh=5248, vv=6400)
  conv2_block(h1s, y2s, whh, b_hh, 0, 3, 4, 2, psum_s, psq_s, wid, lane);
  gn_relu_feat(y2s, 128, 8, gw_hh, gb_hh, psum_s, psq_s, smean, sinv, feat, tile, 3072, tid);
  conv2_block(h1s, y2s, whv, b_hv, 1, 3, 3, 3, psum_s, psq_s, wid, lane);
  gn_relu_feat(y2s, 128, 9, gw_hv, gb_hv, psum_s, psq_s, smean, sinv, feat, tile, 4096, tid);
  conv2_block(v1s, y2s, wvh, b_vh, 0, 4, 3, 3, psum_s, psq_s, wid, lane);
  gn_relu_feat(y2s, 128, 9, gw_vh, gb_vh, psum_s, psq_s, smean, sinv, feat, tile, 5248, tid);
  conv2_block(v1s, y2s, wvv, b_vv, 1, 4, 2, 4, psum_s, psq_s, wid, lane);
  gn_relu_feat(y2s, 128, 8, gw_vv, gb_vv, psum_s, psq_s, smean, sinv, feat, tile, 6400, tid);
}

// ---------------------------------------------------------------------------
// FC layer 1: hidden(B x 512) = ReLU(feat(B x 7424) * [vw1|aw1]^T + bias)
// One workgroup per 16-row M-tile; 8 waves x 4 N-tiles each; K = 232 x 32.
// ---------------------------------------------------------------------------
__global__ __launch_bounds__(256) void mlp1_kernel(
    const _Float16* __restrict__ feat, const _Float16* __restrict__ w1p,
    const float* __restrict__ vb1, const float* __restrict__ ab1,
    float* __restrict__ hidden) {
  const int tid = threadIdx.x, lane = tid & 31, wid = tid >> 5;
  const unsigned mt = blockIdx.x;
  V8 acc[4];
#pragma unroll
  for (int j = 0; j < 4; j++) { v8f z = {}; acc[j].v = z; }

  const _Float16* ap = feat + (size_t)mt * 118784u + (unsigned)lane * 16u; // +512/kt
  for (int kt = 0; kt < 232; kt++) {
    v16h a = *(const v16h*)(ap + (unsigned)kt * 512u);
    __builtin_prefetch(ap + (unsigned)(kt + 1) * 512u, 0, 0);   // global_prefetch_b8
#pragma unroll
    for (int j = 0; j < 4; j++) {
      int nt = wid * 4 + j;
      v16h b = *(const v16h*)(w1p + ((unsigned)kt * 32u + (unsigned)nt) * 512u +
                              (unsigned)lane * 16u);
      acc[j].v = wmma16x16x32(a, b, acc[j].v);
    }
  }
  const int kh2 = lane >> 4;
#pragma unroll
  for (int j = 0; j < 4; j++) {
    int nt = wid * 4 + j;
    int n = nt * 16 + (lane & 15);                 // 0..255 value head, 256..511 adv head
    float bias = (n < 256) ? vb1[n] : ab1[n - 256];
#pragma unroll
    for (int r = 0; r < 8; r++) {
      int s = (int)mt * 16 + r + (kh2 << 3);
      float y = acc[j].e[r] + bias;
      hidden[(size_t)s * 512 + n] = y > 0.f ? y : 0.f;
    }
  }
}

// ---------------------------------------------------------------------------
// Heads: per-sample v + (a - mean(a)); 256-length dots, one thread per sample.
// ---------------------------------------------------------------------------
__global__ void heads_kernel(const float* __restrict__ hidden,
                             const float* __restrict__ vw2, const float* __restrict__ vb2,
                             const float* __restrict__ aw2, const float* __restrict__ ab2,
                             float* __restrict__ out, int B) {
  int s = blockIdx.x * blockDim.x + threadIdx.x;
  if (s >= B) return;
  const float4* hv = (const float4*)(hidden + (size_t)s * 512);
  const float4* ha = hv + 64;
  const float4* w2 = (const float4*)vw2;
  float v = vb2[0];
  for (int j = 0; j < 64; j++) {
    float4 h = hv[j], w = w2[j];
    v += h.x * w.x + h.y * w.y + h.z * w.z + h.w * w.w;
  }
  float a[4], am = 0.f;
#pragma unroll
  for (int k = 0; k < 4; k++) {
    const float4* wa = (const float4*)(aw2 + k * 256);
    float t = ab2[k];
    for (int j = 0; j < 64; j++) {
      float4 h = ha[j], w = wa[j];
      t += h.x * w.x + h.y * w.y + h.z * w.z + h.w * w.w;
    }
    a[k] = t; am += t;
  }
  am *= 0.25f;
#pragma unroll
  for (int k = 0; k < 4; k++) out[(size_t)s * 4 + k] = v + a[k] - am;
}

// ---------------------------------------------------------------------------
extern "C" void kernel_launch(void* const* d_in, const int* in_sizes, int n_in,
                              void* d_out, int out_size, void* d_ws, size_t ws_size,
                              hipStream_t stream) {
  const float* x    = (const float*)d_in[0];
  const float* w_h1 = (const float*)d_in[1];  const float* b_h1 = (const float*)d_in[2];
  const float* w_v1 = (const float*)d_in[3];  const float* b_v1 = (const float*)d_in[4];
  const float* w_hh = (const float*)d_in[5];  const float* b_hh = (const float*)d_in[6];
  const float* w_hv = (const float*)d_in[7];  const float* b_hv = (const float*)d_in[8];
  const float* w_vh = (const float*)d_in[9];  const float* b_vh = (const float*)d_in[10];
  const float* w_vv = (const float*)d_in[11]; const float* b_vv = (const float*)d_in[12];
  const float* gw_h1 = (const float*)d_in[13]; const float* gb_h1 = (const float*)d_in[14];
  const float* gw_v1 = (const float*)d_in[15]; const float* gb_v1 = (const float*)d_in[16];
  const float* gw_hh = (const float*)d_in[17]; const float* gb_hh = (const float*)d_in[18];
  const float* gw_hv = (const float*)d_in[19]; const float* gb_hv = (const float*)d_in[20];
  const float* gw_vh = (const float*)d_in[21]; const float* gb_vh = (const float*)d_in[22];
  const float* gw_vv = (const float*)d_in[23]; const float* gb_vv = (const float*)d_in[24];
  const float* vw1 = (const float*)d_in[25]; const float* vb1 = (const float*)d_in[26];
  const float* vw2 = (const float*)d_in[27]; const float* vb2 = (const float*)d_in[28];
  const float* aw1 = (const float*)d_in[29]; const float* ab1 = (const float*)d_in[30];
  const float* aw2 = (const float*)d_in[31]; const float* ab2 = (const float*)d_in[32];

  const int B = in_sizes[0] / 256;        // 4096
  const int tiles = B / 16;               // 256

  // workspace carve (all 256B aligned)
  char* p = (char*)d_ws;
  _Float16* W1P = (_Float16*)p; p += 7602176;          // 232*32*32*16 f16
  _Float16* WH1 = (_Float16*)p; p += 8192;
  _Float16* WV1 = (_Float16*)p; p += 8192;
  _Float16* WHH = (_Float16*)p; p += 65536;
  _Float16* WHV = (_Float16*)p; p += 65536;
  _Float16* WVH = (_Float16*)p; p += 65536;
  _Float16* WVV = (_Float16*)p; p += 65536;
  _Float16* FEAT = (_Float16*)p; p += (size_t)B * 7424 * 2;
  float*    HID  = (float*)p;                          // B*512 f32

  // weight prep
  {
    int tot = 256 * 7424;
    prep_fc<<<(tot + 255) / 256, 256, 0, stream>>>(vw1, W1P, 0, 256, 7424);
    prep_fc<<<(tot + 255) / 256, 256, 0, stream>>>(aw1, W1P, 16, 256, 7424);
  }
  prep_conv<<<(128 * 16 * 2 + 255) / 256, 256, 0, stream>>>(w_h1, WH1, 16);
  prep_conv<<<(128 * 16 * 2 + 255) / 256, 256, 0, stream>>>(w_v1, WV1, 16);
  prep_conv<<<(128 * 128 * 2 + 255) / 256, 256, 0, stream>>>(w_hh, WHH, 128);
  prep_conv<<<(128 * 128 * 2 + 255) / 256, 256, 0, stream>>>(w_hv, WHV, 128);
  prep_conv<<<(128 * 128 * 2 + 255) / 256, 256, 0, stream>>>(w_vh, WVH, 128);
  prep_conv<<<(128 * 128 * 2 + 255) / 256, 256, 0, stream>>>(w_vv, WVV, 128);

  // conv + GN + ReLU -> feat (dynamic LDS: x + h1 + v1 + y2 + stats = 144.5 KB)
  size_t smem = (size_t)(4096 + 24576 + 24576 + 18432) * 2 + (128 + 128 + 16 + 16) * 4;
  conv_gn_kernel<<<tiles, 256, smem, stream>>>(
      x, b_h1, b_v1, b_hh, b_hv, b_vh, b_vv,
      gw_h1, gb_h1, gw_v1, gb_v1, gw_hh, gb_hh, gw_hv, gb_hv,
      gw_vh, gb_vh, gw_vv, gb_vv,
      WH1, WV1, WHH, WHV, WVH, WVV, FEAT);

  // fused value+advantage FC layer 1
  mlp1_kernel<<<tiles, 256, 0, stream>>>(FEAT, W1P, vb1, ab1, HID);

  // heads + dueling combine
  heads_kernel<<<(B + 255) / 256, 256, 0, stream>>>(HID, vw2, vb2, aw2, ab2,
                                                    (float*)d_out, B);
}
